// GatedAttentionSelector_76493367542316
// MI455X (gfx1250) — matrix operand order repaired and checked by reading, not compiled, not run on hardware
//
#include <hip/hip_runtime.h>
#include <hip/hip_bf16.h>

// ---- problem constants (from reference) ----
#define B_      4
#define N_      16384
#define D_      512
#define H_      256
#define K_SEL   11468          // min(max(int(N*0.7),128),N)
#define N_TOP   10322          // K_SEL - N_RAND
#define N_RAND  1146           // max(int(K_SEL*0.1),1)
#define SEL_ELEMS (B_ * K_SEL * D_)        // 23486464 floats
#define ATTN_OFF  SEL_ELEMS
#define IDX_OFF   (SEL_ELEMS + B_ * N_)
#define RAND_CUTOFF 0x57000000u            // ~0.34*2^32 (overshoot, rank-capped later)

#define ROWS_PER_BLK 32
#define LDS_STRIDE   520   // halves; 1040B row stride -> 4 banks apart per row

typedef __attribute__((ext_vector_type(16))) _Float16 v16h;
typedef __attribute__((ext_vector_type(8)))  _Float16 v8h;
typedef __attribute__((ext_vector_type(8)))  float    v8f;
typedef __attribute__((ext_vector_type(4)))  float    v4f;

// =====================================================================
// Kernel 0: pack a (D,H) f32 weight matrix into f16 WMMA B-fragment
// order: frag[kc][nt][lane][e], K = kc*32 + (lane/16)*16 + e,
// N = nt*16 + lane%16.  One thread per fragment-lane (8192 total).
// =====================================================================
__global__ __launch_bounds__(256) void pack_w_kernel(
    const float* __restrict__ W, _Float16* __restrict__ packed)
{
  const int gid  = blockIdx.x * 256 + threadIdx.x;   // 0..8191
  const int lane = gid & 31;
  const int nt   = (gid >> 5) & 15;
  const int kc   = gid >> 9;
  const int lrow = lane & 15, lhi = lane >> 4;
  v16h v;
  #pragma unroll
  for (int e = 0; e < 16; ++e) {
    const int K = kc * 32 + lhi * 16 + e;
    v[e] = (_Float16)W[(size_t)K * H_ + nt * 16 + lrow];
  }
  ((v16h*)packed)[gid] = v;
}

// =====================================================================
// Kernel 1: fused gated-attention scoring.
// raw[b,n] = sum_h tanh(x.Wv+bv)[h] * sigmoid(x.Wu+bu)[h] * w_att[h] + b_att
// Block = 32 rows of flattened (B*N, D); 8 waves; wave w owns H-col-tiles
// {2w, 2w+1} and both 16-row M-tiles.  A staged once in LDS as f16;
// B fragments are pre-packed coalesced v16h loads.
// =====================================================================
__global__ __launch_bounds__(256) void score_kernel(
    const float* __restrict__ feat,
    const _Float16* __restrict__ pWV, const _Float16* __restrict__ pWU,
    const float* __restrict__ bV, const float* __restrict__ bU,
    const float* __restrict__ watt, const float* __restrict__ batt,
    float* __restrict__ raw)
{
  __shared__ _Float16 atile[ROWS_PER_BLK * LDS_STRIDE];  // 33280 B
  __shared__ float lds_raw[ROWS_PER_BLK];
  const int tid  = threadIdx.x;
  const int lane = tid & 31;
  const int wave = tid >> 5;                 // 0..7
  const int row0 = blockIdx.x * ROWS_PER_BLK;

  if (tid < ROWS_PER_BLK) lds_raw[tid] = 0.0f;

  // ---- cooperative stage: thread -> (row = tid/8, 64-float segment) ----
  {
    const int r = tid >> 3, seg = tid & 7;
    const v4f* __restrict__ src =
        (const v4f*)(feat + (size_t)(row0 + r) * D_ + seg * 64);
    _Float16* dst = atile + r * LDS_STRIDE + seg * 64;
    #pragma unroll
    for (int j = 0; j < 8; ++j) {
      const v4f f0 = src[2 * j], f1 = src[2 * j + 1];
      v8h h;
      #pragma unroll
      for (int e = 0; e < 4; ++e) {
        h[e]     = (_Float16)f0[e];
        h[4 + e] = (_Float16)f1[e];
      }
      *(v8h*)(dst + j * 8) = h;               // ds_store_b128
    }
  }
  __syncthreads();

  const int lrow = lane & 15;   // M (A) / N (B,C)
  const int lhi  = lane >> 4;   // lane-half selects K sub-range

  v8f zero = {};
  v8f accV[2][2] = {{zero, zero}, {zero, zero}};  // [nt_local][m]
  v8f accU[2][2] = {{zero, zero}, {zero, zero}};

  const v16h* __restrict__ fragV = (const v16h*)pWV;
  const v16h* __restrict__ fragU = (const v16h*)pWU;

  for (int kc = 0; kc < D_ / 32; ++kc) {
    if (kc + 1 < D_ / 32)
      __builtin_prefetch(
          (const void*)(fragV + (((size_t)(kc + 1) * 16 + wave * 2) * 32 + lane)),
          0, 0);                               // global_prefetch_b8

    // ---- A fragments (both M-tiles) from LDS: K = lhi*16 + e ----
    v16h am[2];
    #pragma unroll
    for (int m = 0; m < 2; ++m) {
      const _Float16* ap =
          atile + (m * 16 + lrow) * LDS_STRIDE + kc * 32 + lhi * 16;
      const v8h lo = *(const v8h*)ap;          // ds_load_b128
      const v8h hi = *(const v8h*)(ap + 8);    // ds_load_b128
      v16h a;
      #pragma unroll
      for (int e = 0; e < 8; ++e) { a[e] = lo[e]; a[8 + e] = hi[e]; }
      am[m] = a;
    }

    #pragma unroll
    for (int i = 0; i < 2; ++i) {
      const int nt = wave * 2 + i;
      const size_t foff = ((size_t)kc * 16 + nt) * 32 + lane;
      const v16h bv = fragV[foff];             // coalesced global_load_b128 x2
      const v16h bu = fragU[foff];
      #pragma unroll
      for (int m = 0; m < 2; ++m) {
        accV[i][m] = __builtin_amdgcn_wmma_f32_16x16x32_f16(
            false, am[m], false, bv, (short)0, accV[i][m], false, false);
        accU[i][m] = __builtin_amdgcn_wmma_f32_16x16x32_f16(
            false, am[m], false, bu, (short)0, accU[i][m], false, false);
      }
    }
  }

  // ---- gate + reduce over h (C layout: M = r + 8*lhi, N = lane%16) ----
  #pragma unroll
  for (int i = 0; i < 2; ++i) {
    const int h = (wave * 2 + i) * 16 + lrow;
    const float bvv = bV[h], buu = bU[h], wa = watt[h];
    #pragma unroll
    for (int m = 0; m < 2; ++m) {
      #pragma unroll
      for (int r = 0; r < 8; ++r) {
        const int M = r + lhi * 8;
        const float g = tanhf(accV[i][m][r] + bvv) * wa /
                        (1.0f + __expf(-(accU[i][m][r] + buu)));
        atomicAdd(&lds_raw[m * 16 + M], g);    // ds_add_f32
      }
    }
  }
  __syncthreads();
  if (tid < ROWS_PER_BLK) raw[row0 + tid] = lds_raw[tid] + batt[0];
}

// =====================================================================
// Kernel 2: per-batch softmax over N=16384 (one block per batch row)
// =====================================================================
__global__ __launch_bounds__(1024) void softmax_kernel(
    const float* __restrict__ raw, float* __restrict__ attn)
{
  __shared__ float s[1024];
  const int b = blockIdx.x, t = threadIdx.x;
  const float* __restrict__ r = raw + (size_t)b * N_;

  float m = -1e30f;
  for (int i = t; i < N_; i += 1024) m = fmaxf(m, r[i]);
  s[t] = m; __syncthreads();
  for (int off = 512; off > 0; off >>= 1) {
    if (t < off) s[t] = fmaxf(s[t], s[t + off]);
    __syncthreads();
  }
  m = s[0]; __syncthreads();

  float sum = 0.0f;
  for (int i = t; i < N_; i += 1024) sum += __expf(r[i] - m);
  s[t] = sum; __syncthreads();
  for (int off = 512; off > 0; off >>= 1) {
    if (t < off) s[t] += s[t + off];
    __syncthreads();
  }
  const float inv = 1.0f / s[0];
  for (int i = t; i < N_; i += 1024)
    attn[(size_t)b * N_ + i] = __expf(r[i] - m) * inv;
}

// =====================================================================
// Kernel 3: radix threshold for top-N_TOP (attn>=0 -> bits monotone).
// =====================================================================
__global__ __launch_bounds__(1024) void thresh_kernel(
    const float* __restrict__ attn, int* __restrict__ tinfo)
{
  __shared__ unsigned hist[4096];
  const int b = blockIdx.x, t = threadIdx.x;
  for (int i = t; i < 4096; i += 1024) hist[i] = 0u;
  __syncthreads();
  const float* __restrict__ a = attn + (size_t)b * N_;
  for (int i = t; i < N_; i += 1024)
    atomicAdd(&hist[__float_as_uint(a[i]) >> 18], 1u);   // ds_add_u32
  __syncthreads();
  if (t == 0) {
    unsigned cum = 0; int T = 0; unsigned above = 0;
    for (int bin = 4095; bin >= 0; --bin) {
      const unsigned c = hist[bin];
      if (cum + c >= (unsigned)N_TOP) { T = bin; above = cum; break; }
      cum += c;
    }
    tinfo[b * 2 + 0] = T;
    tinfo[b * 2 + 1] = (int)(N_TOP - above);
  }
}

// =====================================================================
// Kernel 4: deterministic selection + ordered compaction (exactly K_SEL).
// =====================================================================
__device__ __forceinline__ int block_exscan(int v, int* sd, int t, int* total) {
  sd[t] = v; __syncthreads();
  for (int off = 1; off < 1024; off <<= 1) {
    int x = sd[t];
    if (t >= off) x += sd[t - off];
    __syncthreads();
    sd[t] = x;
    __syncthreads();
  }
  const int incl = sd[t];
  *total = sd[1023];
  __syncthreads();
  return incl - v;
}

__device__ __forceinline__ unsigned hash_u32(unsigned x) {
  x ^= x >> 16; x *= 0x7feb352dU;
  x ^= x >> 15; x *= 0x846ca68bU;
  x ^= x >> 16; return x;
}

__global__ __launch_bounds__(1024) void select_kernel(
    const float* __restrict__ attn, const int* __restrict__ tinfo,
    int* __restrict__ sel, float* __restrict__ selidx_f)
{
  __shared__ int sd[1024];
  const int b = blockIdx.x, t = threadIdx.x;
  const float* __restrict__ a = attn + (size_t)b * N_;
  const int T = tinfo[b * 2], need_eq = tinfo[b * 2 + 1];
  const int n0 = t * 16;

  unsigned gt = 0, eq = 0;
  #pragma unroll
  for (int e = 0; e < 16; ++e) {
    const int bin = (int)(__float_as_uint(a[n0 + e]) >> 18);
    if (bin > T)       gt |= 1u << e;
    else if (bin == T) eq |= 1u << e;
  }

  int total;
  int base = block_exscan(__popc(eq), sd, t, &total);
  unsigned member = gt;
  { int run = base;
    for (int e = 0; e < 16; ++e)
      if ((eq >> e) & 1) { if (run < need_eq) member |= 1u << e; ++run; } }

  unsigned rc = 0;
  #pragma unroll
  for (int e = 0; e < 16; ++e)
    if (!((member >> e) & 1) &&
        hash_u32(((unsigned)b << 14) | (unsigned)(n0 + e)) < RAND_CUTOFF)
      rc |= 1u << e;
  base = block_exscan(__popc(rc), sd, t, &total);
  { int run = base;
    for (int e = 0; e < 16; ++e)
      if ((rc >> e) & 1) { if (run < N_RAND) member |= 1u << e; ++run; } }

  base = block_exscan(__popc(member), sd, t, &total);
  const int deficit = K_SEL - total;      // uniform (broadcast total)
  if (deficit > 0) {
    const unsigned nm = (~member) & 0xFFFFu;
    int tot2;
    const int b2 = block_exscan(__popc(nm), sd, t, &tot2);
    int run = b2;
    for (int e = 0; e < 16; ++e)
      if ((nm >> e) & 1) { if (run < deficit) member |= 1u << e; ++run; }
  }

  int pos = block_exscan(__popc(member), sd, t, &total);
  for (int e = 0; e < 16; ++e)
    if ((member >> e) & 1) {
      sel[(size_t)b * K_SEL + pos]      = n0 + e;
      selidx_f[(size_t)b * K_SEL + pos] = (float)(n0 + e);
      ++pos;
    }
}

// =====================================================================
// Kernel 5: gather selected rows (pure B128 bandwidth copy).
// =====================================================================
__global__ __launch_bounds__(128) void gather_kernel(
    const float* __restrict__ feat, const int* __restrict__ sel,
    float* __restrict__ out)
{
  const int blk = blockIdx.x;            // 0 .. B_*K_SEL-1
  const int b   = blk / K_SEL;
  const int row = sel[blk];
  const v4f* __restrict__ src =
      (const v4f*)(feat + ((size_t)b * N_ + row) * D_);
  v4f* __restrict__ dst = (v4f*)(out + (size_t)blk * D_);
  dst[threadIdx.x] = src[threadIdx.x];
}

// =====================================================================
extern "C" void kernel_launch(void* const* d_in, const int* in_sizes, int n_in,
                              void* d_out, int out_size, void* d_ws, size_t ws_size,
                              hipStream_t stream) {
  const float* feat = (const float*)d_in[0];
  const float* WV   = (const float*)d_in[1];
  const float* bV   = (const float*)d_in[2];
  const float* WU   = (const float*)d_in[3];
  const float* bU   = (const float*)d_in[4];
  const float* watt = (const float*)d_in[5];
  const float* batt = (const float*)d_in[6];

  float* out      = (float*)d_out;
  float* selected = out;                 // (B, K_SEL, D)
  float* attn     = out + ATTN_OFF;      // (B, N)
  float* selidx   = out + IDX_OFF;       // (B, K_SEL) as float

  // workspace layout (bytes)
  char* ws = (char*)d_ws;
  float*     raw  = (float*)ws;                         // 262144 B
  int*       tinf = (int*)(ws + 262144);                // 64 B
  int*       sel  = (int*)(ws + 262208);                // 183488 B
  _Float16*  pWV  = (_Float16*)(ws + 458752);           // 262144 B
  _Float16*  pWU  = (_Float16*)(ws + 720896);           // 262144 B -> 983040 total

  pack_w_kernel <<<32, 256, 0, stream>>>(WV, pWV);
  pack_w_kernel <<<32, 256, 0, stream>>>(WU, pWU);
  score_kernel  <<<(B_ * N_) / ROWS_PER_BLK, 256, 0, stream>>>(
      feat, pWV, pWU, bV, bU, watt, batt, raw);
  softmax_kernel<<<B_, 1024, 0, stream>>>(raw, attn);
  thresh_kernel <<<B_, 1024, 0, stream>>>(attn, tinf);
  select_kernel <<<B_, 1024, 0, stream>>>(attn, tinf, sel, selidx);
  gather_kernel <<<B_ * K_SEL, 128, 0, stream>>>(feat, sel, selected);
}
